// UniVerseTransformerBlock_17549236372277
// MI455X (gfx1250) — matrix operand order repaired and checked by reading, not compiled, not run on hardware
//
#include <hip/hip_runtime.h>
#include <hip/hip_bf16.h>
#include <stdint.h>

// ---------------------------------------------------------------- constants
#define B_  2
#define S_  4096
#define C_  2048
#define H_  16
#define D_  128
#define M_  (B_ * S_)            // 8192 token rows
#define PADP 144                 // 129 (v_pad) rounded up to 9 WMMA tiles
#define NCHUNK 8                 // S-chunks for partial VK
#define SCHUNK (S_ / NCHUNK)     // 512

typedef __attribute__((ext_vector_type(16))) __bf16  v16bf;
typedef __attribute__((ext_vector_type(8)))  float   v8f;
typedef __attribute__((ext_vector_type(4)))  unsigned int u32x4;
typedef int v4i __attribute__((vector_size(16)));    // async builtin's pointee type

union FragLd { v16bf v; u32x4 q[2]; };
union Bf4Ld  { short4 s4; __bf16 h[4]; };

__device__ inline v8f vzero8() {
  v8f z = {0.f, 0.f, 0.f, 0.f, 0.f, 0.f, 0.f, 0.f};
  return z;
}

__device__ inline v8f wmma_bf16(v16bf a, v16bf b, v8f c) {
  return __builtin_amdgcn_wmma_f32_16x16x32_bf16(
      /*neg_a=*/false, a, /*neg_b=*/false, b,
      /*c_mod=*/(short)0, c, /*reuse_a=*/false, /*reuse_b=*/false);
}

// A fragment (16x32, M x K): lane l in [0,16) -> row M=l, K chunks {half*8..+8, 16+half*8..+8}
__device__ inline v16bf ld_frag_a(const __bf16* tile, int stride, int row0, int kbase) {
  const int l = threadIdx.x & 31;
  const int half = l >> 4;
  const __bf16* rp = tile + (size_t)(row0 + (l & 15)) * stride;
  FragLd u;
  u.q[0] = *(const u32x4*)(rp + kbase + half * 8);
  u.q[1] = *(const u32x4*)(rp + kbase + half * 8 + 16);
  return u.v;
}

// B fragment (32x16, K x N) with tile stored transposed as [n][k]:
// lane l -> column n = l&15, K chunks {half*8..+8, 16+half*8..+8}
__device__ inline v16bf ld_frag_b(const __bf16* tileT, int stride, int row0, int kbase) {
  const int l = threadIdx.x & 31;
  const int half = l >> 4;
  const __bf16* rp = tileT + (size_t)(row0 + (l & 15)) * stride;
  FragLd u;
  u.q[0] = *(const u32x4*)(rp + kbase + half * 8);
  u.q[1] = *(const u32x4*)(rp + kbase + half * 8 + 16);
  return u.v;
}

// -------------------------------------------------- optional async LDS copy
#if defined(__has_builtin)
#if __has_builtin(__builtin_amdgcn_global_load_async_to_lds_b128)
#define HAVE_ASYNC_LDS 1
#endif
#endif

#ifdef HAVE_ASYNC_LDS
#define ASYNC_CP_B128(gptr, lptr) \
  __builtin_amdgcn_global_load_async_to_lds_b128((v4i*)(gptr), (v4i*)(lptr), 0, 0)
#define ASYNC_WAIT_LE4() asm volatile("s_wait_asynccnt 0x4" ::: "memory")
#define ASYNC_WAIT_LE0() asm volatile("s_wait_asynccnt 0x0" ::: "memory")
#else
#define ASYNC_CP_B128(gptr, lptr) (*(u32x4*)(lptr) = *(const u32x4*)(gptr))
#define ASYNC_WAIT_LE4() ((void)0)
#define ASYNC_WAIT_LE0() ((void)0)
#endif

// ---------------------------------------------------------------- kernels --

__global__ void f32_to_bf16_kernel(const float* __restrict__ src,
                                   __bf16* __restrict__ dst, int n) {
  for (int i = blockIdx.x * blockDim.x + threadIdx.x; i < n;
       i += gridDim.x * blockDim.x)
    dst[i] = (__bf16)src[i];
}

// WT[n][k] = (bf16) W[k][n]   (W is K x N row-major)
__global__ __launch_bounds__(256) void transpose_convert_kernel(
    const float* __restrict__ W, __bf16* __restrict__ WT, int K, int N) {
  __shared__ float t[32][33];
  const int x = threadIdx.x & 31, y0 = threadIdx.x >> 5;
  const int n0 = blockIdx.x * 32, k0 = blockIdx.y * 32;
  for (int j = 0; j < 4; ++j) {
    int r = y0 + j * 8;
    t[r][x] = W[(size_t)(k0 + r) * N + n0 + x];
  }
  __syncthreads();
  for (int j = 0; j < 4; ++j) {
    int r = y0 + j * 8;
    WT[(size_t)(n0 + r) * K + k0 + x] = (__bf16)t[x][r];
  }
}

// O[M x N] = A[M x K] * BT[N x K]^T ; bf16 in, f32 accum, double-buffered
// async-LDS pipeline. BF16OUT=true -> bf16 output, else f32 + bias.
template <bool BF16OUT>
__global__ __launch_bounds__(256) void gemm_bf16_kernel(
    const __bf16* __restrict__ A, const __bf16* __restrict__ BT,
    int Mdim, int Ndim, int Kdim,
    __bf16* __restrict__ outB, float* __restrict__ outF,
    const float* __restrict__ bias) {
  __shared__ __bf16 at[2][128][40];   // [buf][m][k], K-tile = 32, pad to 40
  __shared__ __bf16 bt[2][128][40];   // [buf][n][k]
  const int tid = threadIdx.x, lane = tid & 31, wid = tid >> 5;
  const int wm = wid & 3, wn = wid >> 2;          // 4 x 2 wave grid
  const int m0 = blockIdx.y * 128, n0 = blockIdx.x * 128;

  v8f acc[2][4];
  for (int mt = 0; mt < 2; ++mt)
    for (int nt = 0; nt < 4; ++nt) acc[mt][nt] = vzero8();

  // 128 rows x 32 bf16 = 512 x 16B chunks per tile; 2 chunks/thread/tile;
  // 4 async ops per thread per k-tile (2 A + 2 B).
  auto issue = [&](int kt, int buf) {
    for (int j = 0; j < 2; ++j) {
      const int cid = tid + 256 * j;
      const int row = cid >> 2, c16 = (cid & 3) * 8;
      const __bf16* ga = A  + (size_t)(m0 + row) * Kdim + kt * 32 + c16;
      const __bf16* gb = BT + (size_t)(n0 + row) * Kdim + kt * 32 + c16;
      ASYNC_CP_B128(ga, &at[buf][row][c16]);
      ASYNC_CP_B128(gb, &bt[buf][row][c16]);
      __builtin_prefetch(ga + 32, 0, 3);          // speculative, next k-tile
      __builtin_prefetch(gb + 32, 0, 3);
    }
  };

  const int nk = Kdim / 32;
  issue(0, 0);
  for (int kt = 0; kt < nk; ++kt) {
    const int cur = kt & 1;
    if (kt + 1 < nk) {
      issue(kt + 1, cur ^ 1);     // overlap next tile's copy with this compute
      ASYNC_WAIT_LE4();           // in-order: cnt<=4 => current tile landed
    } else {
      ASYNC_WAIT_LE0();
    }
    __syncthreads();

    v16bf af[2], bfr[4];
    for (int mt = 0; mt < 2; ++mt)
      af[mt] = ld_frag_a(&at[cur][0][0], 40, wm * 32 + mt * 16, 0);
    for (int nt = 0; nt < 4; ++nt)
      bfr[nt] = ld_frag_b(&bt[cur][0][0], 40, wn * 64 + nt * 16, 0);
    for (int mt = 0; mt < 2; ++mt)
      for (int nt = 0; nt < 4; ++nt)
        acc[mt][nt] = wmma_bf16(af[mt], bfr[nt], acc[mt][nt]);
    __syncthreads();              // frees buf `cur` for tile kt+2
  }

  const int half = lane >> 4, cN = lane & 15;
  for (int mt = 0; mt < 2; ++mt)
    for (int nt = 0; nt < 4; ++nt)
      for (int i = 0; i < 8; ++i) {
        const int r = m0 + wm * 32 + mt * 16 + i + 8 * half;
        const int c = n0 + wn * 64 + nt * 16 + cN;
        const float v = acc[mt][nt][i];
        if constexpr (BF16OUT) outB[(size_t)r * Ndim + c] = (__bf16)v;
        else                   outF[(size_t)r * Ndim + c] = v + bias[c];
      }
}

// Per-row RMSNorm + gain + RoPE + ReLU (lane owns 4 contiguous d's).
__device__ inline void row_norm_rope_relu(const __bf16* rowp, const float* g,
                                          const float* cosb, const float* sinb,
                                          size_t s, int lane, float out[4],
                                          bool do_relu) {
  Bf4Ld r;
  r.s4 = *(const short4*)(rowp + 4 * lane);
  float f[4];
  for (int j = 0; j < 4; ++j) f[j] = (float)r.h[j];
  float ss = f[0] * f[0] + f[1] * f[1] + f[2] * f[2] + f[3] * f[3];
  for (int off = 16; off > 0; off >>= 1) ss += __shfl_xor(ss, off, 32);
  const float inv = rsqrtf(ss * (1.f / (float)D_) + 1e-6f);
  const int d0 = 4 * lane;
  float n[4];
  for (int j = 0; j < 4; ++j) n[j] = f[j] * inv * g[d0 + j];
  const size_t cs = s * D_ + d0;
  const float c0 = cosb[cs], c1 = cosb[cs + 1], c2 = cosb[cs + 2], c3 = cosb[cs + 3];
  const float s0 = sinb[cs], s1 = sinb[cs + 1], s2 = sinb[cs + 2], s3 = sinb[cs + 3];
  out[0] = n[0] * c0 - n[1] * s0;
  out[1] = n[1] * c1 + n[0] * s1;
  out[2] = n[2] * c2 - n[3] * s2;
  out[3] = n[3] * c3 + n[2] * s3;
  if (do_relu)
    for (int j = 0; j < 4; ++j) out[j] = fmaxf(out[j], 0.f);
}

// Partial VK = v_pad^T * kf over one S-chunk. grid (B*H, NCHUNK), 288 thr.
__global__ __launch_bounds__(288) void vk_partial_kernel(
    const __bf16* __restrict__ kb, const __bf16* __restrict__ vb,
    const float* __restrict__ gk, const float* __restrict__ cosb,
    const float* __restrict__ sinb, float* __restrict__ vkpart) {
  extern __shared__ char smem_raw[];
  __bf16* kfT = (__bf16*)smem_raw;          // [D_][136]   (kf transposed: [d][s])
  __bf16* vT  = kfT + D_ * 136;             // [PADP][136] (v_pad transposed: [p][s])
  const int tid = threadIdx.x, lane = tid & 31, wid = tid >> 5;
  const int bh = blockIdx.x, chunk = blockIdx.y;
  const int b = bh >> 4, h = bh & 15;

  // zero pad rows p = 129..143 once
  for (int i = tid; i < (PADP - 129) * 136; i += 288)
    vT[(129 + i / 136) * 136 + (i % 136)] = (__bf16)0.f;

  v8f acc[8];
  for (int dt = 0; dt < 8; ++dt) acc[dt] = vzero8();

  for (int st = 0; st < SCHUNK / 128; ++st) {
    const int sbase = chunk * SCHUNK + st * 128;
    __syncthreads();
    for (int sl = wid; sl < 128; sl += 9) {
      const size_t s = sbase + sl;
      const size_t g = ((size_t)b * S_ + s) * C_ + (size_t)h * D_;
      float kf[4];
      row_norm_rope_relu(kb + g, gk, cosb, sinb, s, lane, kf, true);
      const int d0 = 4 * lane;
      for (int j = 0; j < 4; ++j) kfT[(d0 + j) * 136 + sl] = (__bf16)kf[j];
      Bf4Ld vr;
      vr.s4 = *(const short4*)(vb + g + d0);
      for (int j = 0; j < 4; ++j) vT[(d0 + j) * 136 + sl] = vr.h[j];
      if (lane == 0) vT[128 * 136 + sl] = (__bf16)1.0f;   // pad column = 1
    }
    __syncthreads();
    for (int ks = 0; ks < 4; ++ks) {
      // batch all fragment loads ahead of the WMMA chain (one dscnt wait)
      const v16bf af = ld_frag_a(vT, 136, wid * 16, ks * 32);   // A: p x s
      v16bf bf8[8];
      for (int dt = 0; dt < 8; ++dt)
        bf8[dt] = ld_frag_b(kfT, 136, dt * 16, ks * 32);        // B: s x d
      for (int dt = 0; dt < 8; ++dt)
        acc[dt] = wmma_bf16(af, bf8[dt], acc[dt]);
    }
  }

  float* outp = vkpart + ((size_t)bh * NCHUNK + chunk) * PADP * D_;
  const int half = lane >> 4, cN = lane & 15;
  for (int dt = 0; dt < 8; ++dt)
    for (int i = 0; i < 8; ++i) {
      const int p = wid * 16 + i + 8 * half;
      const int d = dt * 16 + cN;
      outp[(size_t)p * D_ + d] = acc[dt][i];
    }
}

__global__ void vk_reduce_kernel(const float* __restrict__ part,
                                 __bf16* __restrict__ vkb) {
  const int i = blockIdx.x * 256 + threadIdx.x;
  const int TS = PADP * D_;
  if (i >= B_ * H_ * TS) return;
  const int bh = i / TS, r = i % TS;
  float sum = 0.f;
  for (int c = 0; c < NCHUNK; ++c)
    sum += part[((size_t)bh * NCHUNK + c) * TS + r];
  vkb[i] = (__bf16)sum;
}

// out = qf * VK^T, divide by p=128 column, write bf16. grid (B*H, S_/128).
__global__ __launch_bounds__(288) void attn_phase2_kernel(
    const __bf16* __restrict__ qb, const __bf16* __restrict__ vkb,
    const float* __restrict__ gq, const float* __restrict__ cosb,
    const float* __restrict__ sinb, __bf16* __restrict__ attn) {
  extern __shared__ char smem_raw[];
  __bf16* vk_l = (__bf16*)smem_raw;          // [PADP][136]  ([p][d] = B^T layout)
  __bf16* q_t  = vk_l + PADP * 136;          // [128][136]   ([s][d] = A layout)
  float*  o_l  = (float*)(q_t + 128 * 136);  // [128][145]
  const int tid = threadIdx.x, lane = tid & 31, wid = tid >> 5;
  const int bh = blockIdx.x, s0 = blockIdx.y * 128;
  const int b = bh >> 4, h = bh & 15;

  for (int i = tid; i < PADP * D_; i += 288)
    vk_l[(i >> 7) * 136 + (i & 127)] = vkb[(size_t)bh * PADP * D_ + i];

  for (int sl = wid; sl < 128; sl += 9) {
    const size_t s = s0 + sl;
    const size_t g = ((size_t)b * S_ + s) * C_ + (size_t)h * D_;
    float qf[4];
    row_norm_rope_relu(qb + g, gq, cosb, sinb, s, lane, qf, true);
    Bf4Ld w;
    for (int j = 0; j < 4; ++j) w.h[j] = (__bf16)qf[j];
    *(short4*)&q_t[sl * 136 + 4 * lane] = w.s4;
  }
  __syncthreads();

  // B-fragments (VK) depend only on (wid, ks): load once, reuse for all 8 m-tiles.
  v16bf bfr4[4];
  for (int ks = 0; ks < 4; ++ks)
    bfr4[ks] = ld_frag_b(vk_l, 136, wid * 16, ks * 32);          // B: d x p

  const int half = lane >> 4, cN = lane & 15;
  for (int mt = 0; mt < 8; ++mt) {
    v8f acc = vzero8();
    for (int ks = 0; ks < 4; ++ks) {
      const v16bf af = ld_frag_a(q_t, 136, mt * 16, ks * 32);    // A: s x d
      acc = wmma_bf16(af, bfr4[ks], acc);
    }
    for (int i = 0; i < 8; ++i)
      o_l[(mt * 16 + i + 8 * half) * 145 + (wid * 16 + cN)] = acc[i];
  }
  __syncthreads();

  for (int idx = tid; idx < 128 * 128; idx += 288) {
    const int sl = idx >> 7, dd = idx & 127;
    const float den = o_l[sl * 145 + 128] + 1e-15f;
    attn[((size_t)b * S_ + s0 + sl) * C_ + (size_t)h * D_ + dd] =
        (__bf16)(o_l[sl * 145 + dd] / den);
  }
}

// --------------------------------------------------------------- launcher --
extern "C" void kernel_launch(void* const* d_in, const int* in_sizes, int n_in,
                              void* d_out, int out_size, void* d_ws, size_t ws_size,
                              hipStream_t stream) {
  const float* x    = (const float*)d_in[0];
  const float* Wq   = (const float*)d_in[1];
  const float* Wk   = (const float*)d_in[2];
  const float* Wv   = (const float*)d_in[3];
  const float* Wo   = (const float*)d_in[4];
  const float* bo   = (const float*)d_in[5];
  const float* gq   = (const float*)d_in[6];
  const float* gk   = (const float*)d_in[7];
  const float* cosb = (const float*)d_in[8];
  const float* sinb = (const float*)d_in[9];
  float* out = (float*)d_out;

  // workspace carving (256B aligned)
  uintptr_t base = (uintptr_t)d_ws;
  auto alloc = [&](size_t bytes) {
    base = (base + 255) & ~(uintptr_t)255;
    void* p = (void*)base;
    base += bytes;
    return p;
  };
  __bf16* xb   = (__bf16*)alloc((size_t)M_ * C_ * 2);
  __bf16* wqT  = (__bf16*)alloc((size_t)C_ * C_ * 2);
  __bf16* wkT  = (__bf16*)alloc((size_t)C_ * C_ * 2);
  __bf16* wvT  = (__bf16*)alloc((size_t)C_ * C_ * 2);
  __bf16* woT  = (__bf16*)alloc((size_t)C_ * C_ * 2);
  __bf16* qb   = (__bf16*)alloc((size_t)M_ * C_ * 2);
  __bf16* kb   = (__bf16*)alloc((size_t)M_ * C_ * 2);
  __bf16* vb   = (__bf16*)alloc((size_t)M_ * C_ * 2);
  __bf16* attn = (__bf16*)alloc((size_t)M_ * C_ * 2);
  float*  vkp  = (float*)alloc((size_t)B_ * H_ * NCHUNK * PADP * D_ * 4);
  __bf16* vkb  = (__bf16*)alloc((size_t)B_ * H_ * PADP * D_ * 2);
  (void)ws_size; (void)in_sizes; (void)n_in; (void)out_size;

  const size_t smemA = (size_t)(D_ * 136 + PADP * 136) * 2;                  // 73984
  const size_t smemB = (size_t)(PADP * 136 + 128 * 136) * 2 + 128 * 145 * 4; // 148224
  (void)hipFuncSetAttribute((const void*)vk_partial_kernel,
                            hipFuncAttributeMaxDynamicSharedMemorySize, (int)smemA);
  (void)hipFuncSetAttribute((const void*)attn_phase2_kernel,
                            hipFuncAttributeMaxDynamicSharedMemorySize, (int)smemB);

  f32_to_bf16_kernel<<<4096, 256, 0, stream>>>(x, xb, M_ * C_);
  dim3 tg(C_ / 32, C_ / 32);
  transpose_convert_kernel<<<tg, 256, 0, stream>>>(Wq, wqT, C_, C_);
  transpose_convert_kernel<<<tg, 256, 0, stream>>>(Wk, wkT, C_, C_);
  transpose_convert_kernel<<<tg, 256, 0, stream>>>(Wv, wvT, C_, C_);
  transpose_convert_kernel<<<tg, 256, 0, stream>>>(Wo, woT, C_, C_);

  dim3 gg(C_ / 128, M_ / 128);
  gemm_bf16_kernel<true><<<gg, 256, 0, stream>>>(xb, wqT, M_, C_, C_, qb, nullptr, nullptr);
  gemm_bf16_kernel<true><<<gg, 256, 0, stream>>>(xb, wkT, M_, C_, C_, kb, nullptr, nullptr);
  gemm_bf16_kernel<true><<<gg, 256, 0, stream>>>(xb, wvT, M_, C_, C_, vb, nullptr, nullptr);

  vk_partial_kernel<<<dim3(B_ * H_, NCHUNK), 288, smemA, stream>>>(
      kb, vb, gk, cosb, sinb, vkp);
  vk_reduce_kernel<<<(B_ * H_ * PADP * D_ + 255) / 256, 256, 0, stream>>>(vkp, vkb);
  attn_phase2_kernel<<<dim3(B_ * H_, S_ / 128), 288, smemB, stream>>>(
      qb, vkb, gq, cosb, sinb, attn);

  gemm_bf16_kernel<false><<<gg, 256, 0, stream>>>(attn, woT, M_, C_, C_, nullptr, out, bo);
}
